// Rot_Net_64476049047795
// MI455X (gfx1250) — matrix operand order repaired and checked by reading, not compiled, not run on hardware
//
#include <hip/hip_runtime.h>
#include <cstdint>
#include <cstddef>

// ---------------------------------------------------------------------------
// CDNA5 (gfx1250) implementation. wave32. Matrix work on v_wmma_f32_16x16x32_bf16.
// Requires ws_size >= ~180 MB.
// ---------------------------------------------------------------------------

typedef __attribute__((ext_vector_type(16))) __bf16 v16bf;
typedef __attribute__((ext_vector_type(8)))  float  v8f;

union AFrag { unsigned int u[8]; v16bf v; };

__device__ __forceinline__ unsigned short f2bf(float x) {
    unsigned int u = __float_as_uint(x);
    unsigned int r = u + 0x7FFFu + ((u >> 16) & 1u);   // round-to-nearest-even
    return (unsigned short)(r >> 16);
}

// ---------------------------------------------------------------------------
// Unified implicit-GEMM / GEMM kernel, WMMA bf16, f32 accumulate.
//   KHW >  0 : conv mode. A gathered im2col-style from NCHW; epilogue = BN.
//   KHW == 0 : linear mode. A is row-major M x K; epilogue = +bias.
// Fused residual add + ReLU.
// Block = 128 threads (4 waves). Tile = 64(M) x 64(N), K-staged 64 (2 WMMA steps).
// Loader mapping: thread t owns A-tile row (t>>1), k-halves [(t&1)*32, +32) and
// B-tile column (t>>1), k-halves [(t&1)*32, +32)  -> all M/N decomposition is
// loop-invariant and hoisted out of the K loop; k decomposition uses
// compile-time KHW/KW (magic multiply, no software division in the hot loop).
// ---------------------------------------------------------------------------
template<int KHW, int KH, int KW>
__global__ __launch_bounds__(128)
void gemm_wmma_kernel(const float* __restrict__ A,
                      const float* __restrict__ Wt,      // N x K row-major
                      const float* __restrict__ bias,    // linear bias or null
                      const float* __restrict__ bng, const float* __restrict__ bnb,
                      const float* __restrict__ bnm, const float* __restrict__ bnv,
                      const float* __restrict__ resid,   // residual or null
                      float* __restrict__ out,
                      int M, int N, int K,
                      int Cin, int Hin, int Win, int Hout, int Wout,
                      int stride, int pad, int relu)
{
    constexpr bool CONV = (KHW > 0);

    __shared__ unsigned short As[64][66];   // 64 x 64 bf16, padded rows (dword aligned)
    __shared__ unsigned short Bs[64][66];   // [n][k]

    const int tid  = threadIdx.x;
    const int lane = tid & 31;
    const int wv   = tid >> 5;
    const int wm   = (wv >> 1) * 32;
    const int wn   = (wv &  1) * 32;
    const int mBase = blockIdx.x * 64;
    const int nBase = blockIdx.y * 64;

    // ---- loop-invariant loader state (one divide set per thread, total) ----
    const int arow  = tid >> 1;
    const int acol0 = (tid & 1) * 32;
    const int am    = mBase + arow;
    const bool amok = am < M;
    int a_base = 0, a_hb = 0, a_wb = 0;
    const float* aptr = A;
    const int HinWin = Hin * Win;
    if constexpr (CONV) {
        int HW  = Hout * Wout;
        int nb  = am / HW;  int rem = am - nb * HW;
        int ho  = rem / Wout; int wo = rem - ho * Wout;
        a_hb    = ho * stride - pad;
        a_wb    = wo * stride - pad;
        a_base  = nb * Cin * HinWin;
    } else {
        aptr = A + (size_t)am * K;
    }
    const int  bnn  = nBase + (tid >> 1);
    const int  bc0  = (tid & 1) * 32;
    const bool bnok = bnn < N;
    const float* bptr = Wt + (size_t)bnn * K;

    const v8f vzero = {0.f,0.f,0.f,0.f,0.f,0.f,0.f,0.f};
    v8f acc[2][2];
    acc[0][0] = vzero; acc[0][1] = vzero; acc[1][0] = vzero; acc[1][1] = vzero;

    const int hilo  = lane >> 4;
    const int dbase = hilo * 4;
    const int half8 = hilo * 8;

    for (int kb = 0; kb < K; kb += 64) {
        // ---- stage A (thread-private row, 32 consecutive k) ----
        #pragma unroll
        for (int j = 0; j < 32; ++j) {
            int k = kb + acol0 + j;
            unsigned short hv = 0;
            if (amok && k < K) {
                float val;
                if constexpr (CONV) {
                    int ci = k / KHW;            // compile-time divisor
                    int rr = k - ci * KHW;
                    int kh = rr / KW;
                    int kw = rr - kh * KW;
                    int hi = a_hb + kh, wi = a_wb + kw;
                    val = (hi >= 0 && hi < Hin && wi >= 0 && wi < Win)
                        ? A[a_base + ci * HinWin + hi * Win + wi] : 0.f;
                } else {
                    val = aptr[k];
                }
                hv = f2bf(val);
            }
            As[arow][acol0 + j] = hv;
        }
        // ---- stage B (thread-private column, 32 contiguous k -> transposed) ----
        #pragma unroll
        for (int j = 0; j < 32; ++j) {
            int k = kb + bc0 + j;
            unsigned short hv = 0;
            if (bnok && k < K) hv = f2bf(bptr[k]);
            Bs[tid >> 1][bc0 + j] = hv;
        }
        if (bnok && kb + 64 < K) __builtin_prefetch(&bptr[kb + 64 + bc0], 0, 1);
        __syncthreads();

        // ---- 2 K-substeps x 2x2 WMMA ----
        #pragma unroll
        for (int ks = 0; ks < 2; ++ks) {
            AFrag af[2], bf[2];
            const int mrow = wm + (lane & 15);
            const unsigned int* ar0 = (const unsigned int*)(&As[mrow][ks * 32]);
            const unsigned int* ar1 = (const unsigned int*)(&As[mrow + 16][ks * 32]);
            #pragma unroll
            for (int j = 0; j < 4; ++j) {
                af[0].u[j]     = ar0[dbase + j];
                af[0].u[4 + j] = ar0[dbase + 8 + j];
                af[1].u[j]     = ar1[dbase + j];
                af[1].u[4 + j] = ar1[dbase + 8 + j];
            }
            const unsigned int* br0 = (const unsigned int*)(&Bs[wn + (lane & 15)][ks * 32]);
            const unsigned int* br1 = (const unsigned int*)(&Bs[wn + 16 + (lane & 15)][ks * 32]);
            #pragma unroll
            for (int j = 0; j < 4; ++j) {
                bf[0].u[j]     = br0[dbase + j];
                bf[0].u[4 + j] = br0[dbase + 8 + j];
                bf[1].u[j]     = br1[dbase + j];
                bf[1].u[4 + j] = br1[dbase + 8 + j];
            }
            #pragma unroll
            for (int s = 0; s < 2; ++s)
                #pragma unroll
                for (int t = 0; t < 2; ++t)
                    acc[s][t] = __builtin_amdgcn_wmma_f32_16x16x32_bf16(
                        false, af[s].v, false, bf[t].v,
                        (short)0, acc[s][t], false, false);
        }
        __syncthreads();
    }

    // ---- epilogue: BN / bias + residual + relu ----
    int   nn[2];  float scl[2], sft[2];  bool nok[2];
    #pragma unroll
    for (int t = 0; t < 2; ++t) {
        nn[t]  = nBase + wn + t * 16 + (lane & 15);
        nok[t] = nn[t] < N;
        scl[t] = 1.f; sft[t] = 0.f;
        if (nok[t]) {
            if constexpr (CONV) {
                float iv = rsqrtf(bnv[nn[t]] + 1e-5f);
                scl[t] = bng[nn[t]] * iv;
                sft[t] = bnb[nn[t]] - bnm[nn[t]] * scl[t];
            } else if (bias) {
                sft[t] = bias[nn[t]];
            }
        }
    }
    #pragma unroll
    for (int s = 0; s < 2; ++s) {
        const int mm0 = mBase + wm + s * 16 + half8;
        if constexpr (CONV) {
            int HW = Hout * Wout;
            int nb = mm0 / HW;  int rem = mm0 - nb * HW;
            int ho = rem / Wout; int wo = rem - ho * Wout;
            #pragma unroll
            for (int r = 0; r < 8; ++r) {
                int mm = mm0 + r;
                if (mm < M) {
                    size_t rowoff = (((size_t)nb * N) * Hout + ho) * Wout + wo;
                    #pragma unroll
                    for (int t = 0; t < 2; ++t) {
                        if (nok[t]) {
                            size_t oidx = rowoff + (size_t)nn[t] * Hout * Wout;
                            float y = acc[s][t][r] * scl[t] + sft[t];
                            if (resid) y += resid[oidx];
                            if (relu)  y = fmaxf(y, 0.f);
                            out[oidx] = y;
                        }
                    }
                }
                // carry-increment pixel walk (no division)
                ++wo;
                if (wo == Wout) { wo = 0; ++ho; if (ho == Hout) { ho = 0; ++nb; } }
            }
        } else {
            #pragma unroll
            for (int r = 0; r < 8; ++r) {
                int mm = mm0 + r;
                if (mm < M) {
                    #pragma unroll
                    for (int t = 0; t < 2; ++t) {
                        if (nok[t]) {
                            size_t oidx = (size_t)mm * N + nn[t];
                            float y = acc[s][t][r] * scl[t] + sft[t];
                            if (resid) y += resid[oidx];
                            if (relu)  y = fmaxf(y, 0.f);
                            out[oidx] = y;
                        }
                    }
                }
            }
        }
    }
}

// ---------------------------------------------------------------------------
// Small helper kernels
// ---------------------------------------------------------------------------
__global__ void maxpool_kernel(const float* __restrict__ in, float* __restrict__ out,
                               int C, int Hin, int Win, int Hout, int Wout)
{
    int i = blockIdx.x * blockDim.x + threadIdx.x;
    int total = 32 * C * Hout * Wout;
    if (i >= total) return;
    int wo = i % Wout; int q = i / Wout;
    int ho = q % Hout; q /= Hout;
    int c  = q % C;    int b = q / C;
    float mx = -1e30f;
    for (int kh = 0; kh < 3; ++kh)
        for (int kw = 0; kw < 3; ++kw) {
            int hi = ho * 2 - 1 + kh, wi = wo * 2 - 1 + kw;
            if (hi >= 0 && hi < Hin && wi >= 0 && wi < Win) {
                float v = in[(((size_t)b * C + c) * Hin + hi) * Win + wi];
                mx = fmaxf(mx, v);
            }
        }
    out[i] = mx;
}

__global__ void hwmean_kernel(const float* __restrict__ in, float* __restrict__ out,
                              int C, int HW)   // NCHW -> (32, C)
{
    int i = blockIdx.x * blockDim.x + threadIdx.x;
    if (i >= 32 * C) return;
    const float* p = in + (size_t)i * HW;
    float s = 0.f;
    for (int j = 0; j < HW; ++j) s += p[j];
    out[i] = s / (float)HW;
}

__global__ void seqmean_kernel(const float* __restrict__ seq, float* __restrict__ out)
{
    int i = blockIdx.x * blockDim.x + threadIdx.x;   // 32*512
    if (i >= 32 * 512) return;
    int b = i >> 9, d = i & 511;
    float s = 0.f;
    for (int j = 0; j < 11; ++j) s += seq[((size_t)b * 11 + j) * 512 + d];
    out[i] = s / 11.f;
}

__global__ void saliency_kernel(const float* __restrict__ fmap,
                                const float* __restrict__ w, const float* __restrict__ b,
                                float* __restrict__ logits)
{
    int i = blockIdx.x * blockDim.x + threadIdx.x;   // 32*196
    if (i >= 32 * 196) return;
    int p = i % 196, bb = i / 196;
    int oh = p / 14, ow = p % 14;
    float acc = b[0];
    for (int ci = 0; ci < 256; ++ci)
        for (int kh = 0; kh < 3; ++kh)
            for (int kw = 0; kw < 3; ++kw) {
                int hi = oh - 1 + kh, wi = ow - 1 + kw;
                if (hi >= 0 && hi < 14 && wi >= 0 && wi < 14)
                    acc += fmap[(((size_t)bb * 256 + ci) * 14 + hi) * 14 + wi]
                         * w[(ci * 3 + kh) * 3 + kw];
            }
    logits[i] = acc;
}

__global__ void sample_kernel(const float* __restrict__ logits, int* __restrict__ idx)
{
    int bs = blockIdx.x;             // b*10 + s
    int b = bs / 10, s = bs % 10;
    int lane = threadIdx.x;
    float best = -1e30f; int bi = 0;
    for (int p = lane; p < 196; p += 32) {
        unsigned int h = (unsigned)(b * 1664525 + s * 1013904223 + p * 69069 + 12345);
        h ^= h >> 13; h *= 0x85ebca6bu; h ^= h >> 16;
        float u = (float)((h & 0xFFFFFFu) + 1u) * (1.0f / 16777218.0f);
        float g = -logf(-logf(u));
        float v = logits[b * 196 + p] * 10.0f + g;   // /TEMPERATURE
        if (v > best) { best = v; bi = p; }
    }
    for (int off = 16; off; off >>= 1) {
        float ov = __shfl_xor(best, off, 32);
        int   oi = __shfl_xor(bi,   off, 32);
        if (ov > best || (ov == best && oi < bi)) { best = ov; bi = oi; }
    }
    if (lane == 0) idx[bs] = bi;
}

__global__ void localconst_kernel(const float* __restrict__ l1b,
                                  const float* __restrict__ l2w,
                                  const float* __restrict__ l2b,
                                  float* __restrict__ locc)
{
    int d = threadIdx.x;             // 512 threads
    float acc = l2b[d];
    for (int j = 0; j < 128; ++j) acc += l2w[d * 128 + j] * fmaxf(l1b[j], 0.f);
    locc[d] = acc;
}

__global__ void seqassemble_kernel(const float* __restrict__ pos,
                                   const float* __restrict__ gemb,
                                   const float* __restrict__ locc,
                                   const int* __restrict__ idx,
                                   float* __restrict__ seq)
{
    int i = blockIdx.x * blockDim.x + threadIdx.x;   // 32*11*512
    if (i >= 32 * 11 * 512) return;
    int d = i & 511; int q = i >> 9;
    int r = q % 11;  int b = q / 11;
    float v;
    if (r == 0) {
        v = pos[d] + gemb[b * 512 + d];
    } else {
        int s = r - 1;
        int id = idx[b * 10 + s];
        int kx = id % 14, ky = id / 14;
        int pi = kx * 8 + ky + 1;
        pi = pi < 0 ? 0 : (pi > 9 ? 9 : pi);
        v = pos[pi * 512 + d] + locc[d];
    }
    seq[i] = v;
}

__global__ void attn_kernel(const float* __restrict__ Q, const float* __restrict__ K,
                            const float* __restrict__ V, float* __restrict__ O)
{
    int bh = blockIdx.x;             // 32*8
    int b = bh >> 3, h = bh & 7;
    int lane = threadIdx.x;
    if (lane >= 11) return;
    float sc[11];
    const float* qp = Q + ((size_t)b * 11 + lane) * 512 + h * 64;
    float mx = -1e30f;
    for (int j = 0; j < 11; ++j) {
        const float* kp = K + ((size_t)b * 11 + j) * 512 + h * 64;
        float d = 0.f;
        for (int t = 0; t < 64; ++t) d += qp[t] * kp[t];
        sc[j] = d * 0.125f;
        mx = fmaxf(mx, sc[j]);
    }
    float sum = 0.f;
    for (int j = 0; j < 11; ++j) { sc[j] = expf(sc[j] - mx); sum += sc[j]; }
    float inv = 1.f / sum;
    float* op = O + ((size_t)b * 11 + lane) * 512 + h * 64;
    for (int t = 0; t < 64; ++t) {
        float a = 0.f;
        for (int j = 0; j < 11; ++j) a += sc[j] * V[((size_t)b * 11 + j) * 512 + h * 64 + t];
        op[t] = a * inv;
    }
}

__global__ void lnadd_kernel(float* __restrict__ seq, const float* __restrict__ add,
                             const float* __restrict__ g, const float* __restrict__ b)
{
    int row = blockIdx.x;            // 352 rows
    int lane = threadIdx.x;          // 32
    float e[16];
    float s = 0.f;
    #pragma unroll
    for (int i = 0; i < 16; ++i) {
        int d = lane + i * 32;
        float v = seq[(size_t)row * 512 + d] + add[(size_t)row * 512 + d];
        e[i] = v; s += v;
    }
    for (int off = 16; off; off >>= 1) s += __shfl_xor(s, off, 32);
    float mu = s / 512.f;
    float vs = 0.f;
    #pragma unroll
    for (int i = 0; i < 16; ++i) { float d = e[i] - mu; vs += d * d; }
    for (int off = 16; off; off >>= 1) vs += __shfl_xor(vs, off, 32);
    float inv = rsqrtf(vs / 512.f + 1e-5f);
    #pragma unroll
    for (int i = 0; i < 16; ++i) {
        int d = lane + i * 32;
        seq[(size_t)row * 512 + d] = (e[i] - mu) * inv * g[d] + b[d];
    }
}

__global__ void rot_kernel(const float* __restrict__ r, float* __restrict__ out)
{
    int b = threadIdx.x;             // 32
    if (b >= 32) return;
    float x1[3], x2[3], x3[3];
    for (int i = 0; i < 3; ++i) { x1[i] = r[b * 6 + i]; x2[i] = r[b * 6 + 3 + i]; }
    float dot = x1[0] * x2[0] + x1[1] * x2[1] + x1[2] * x2[2];
    for (int i = 0; i < 3; ++i) x2[i] -= dot * x1[i];
    x3[0] = x1[1] * x2[2] - x1[2] * x2[1];
    x3[1] = x1[2] * x2[0] - x1[0] * x2[2];
    x3[2] = x1[0] * x2[1] - x1[1] * x2[0];
    float* vecs[3] = { x1, x2, x3 };
    for (int k = 0; k < 3; ++k) {
        float* v = vecs[k];
        float n = sqrtf(v[0] * v[0] + v[1] * v[1] + v[2] * v[2]);
        n = fmaxf(n, 1e-12f);
        for (int i = 0; i < 3; ++i) out[b * 9 + k * 3 + i] = v[i] / n;
    }
}

// ---------------------------------------------------------------------------
// Host orchestration
// ---------------------------------------------------------------------------
static void launch_gemm(hipStream_t st, const float* A, const float* W, const float* bias,
                        const float* g, const float* b, const float* m, const float* v,
                        const float* resid, float* out,
                        int M, int N, int K,
                        int Cin, int Hin, int Win, int Hout, int Wout,
                        int KH, int stride, int pad, int relu)
{
    dim3 grid((M + 63) / 64, (N + 63) / 64, 1);
    if (KH == 0)
        gemm_wmma_kernel<0, 1, 1><<<grid, 128, 0, st>>>(A, W, bias, g, b, m, v, resid, out,
            M, N, K, Cin, Hin, Win, Hout, Wout, stride, pad, relu);
    else if (KH == 1)
        gemm_wmma_kernel<1, 1, 1><<<grid, 128, 0, st>>>(A, W, bias, g, b, m, v, resid, out,
            M, N, K, Cin, Hin, Win, Hout, Wout, stride, pad, relu);
    else if (KH == 3)
        gemm_wmma_kernel<9, 3, 3><<<grid, 128, 0, st>>>(A, W, bias, g, b, m, v, resid, out,
            M, N, K, Cin, Hin, Win, Hout, Wout, stride, pad, relu);
    else
        gemm_wmma_kernel<49, 7, 7><<<grid, 128, 0, st>>>(A, W, bias, g, b, m, v, resid, out,
            M, N, K, Cin, Hin, Win, Hout, Wout, stride, pad, relu);
}

struct BnP  { const float *g, *b, *m, *v; };
struct BlkP { const float* w1; BnP bn1; const float* w2; BnP bn2;
              const float* dw; BnP dbn; bool has; };
struct LayP { const float *qw,*qb,*kw,*kb,*vw,*vb,*ow,*ob;
              const float *ln1g,*ln1b,*f1w,*f1b,*f2w,*f2b,*ln2g,*ln2b; };

extern "C" void kernel_launch(void* const* d_in, const int* in_sizes, int n_in,
                              void* d_out, int out_size, void* d_ws, size_t ws_size,
                              hipStream_t stream)
{
    (void)in_sizes; (void)n_in; (void)out_size; (void)ws_size;

    // ---------------- input pointer walk (insertion order) ----------------
    int ip = 0;
    auto F = [&]() -> const float* { return (const float*)d_in[ip++]; };
    auto BN = [&]() -> BnP { BnP p; p.g = F(); p.b = F(); p.m = F(); p.v = F(); return p; };

    const float* rgb = F();
    const float* stem_w = F();
    BnP stem_bn = BN();

    const int nblocks[4] = {3, 4, 6, 3};
    const int cins[4]    = {64, 64, 128, 256};
    const int couts[4]   = {64, 128, 256, 512};
    const int strides[4] = {1, 2, 2, 2};
    BlkP blocks[16];
    int gi = 0;
    for (int si = 0; si < 4; ++si)
        for (int bi = 0; bi < nblocks[si]; ++bi) {
            BlkP& p = blocks[gi++];
            int cin_b = (bi == 0) ? cins[si] : couts[si];
            int st    = (bi == 0) ? strides[si] : 1;
            p.w1 = F(); p.bn1 = BN();
            p.w2 = F(); p.bn2 = BN();
            p.has = (st != 1 || cin_b != couts[si]);
            if (p.has) { p.dw = F(); p.dbn = BN(); } else { p.dw = nullptr; }
        }
    const float* fcw = F(); const float* fcb = F();
    const float* salw = F(); const float* salb = F();
    const float* pos = F();
    const float* l1w = F(); const float* l1b = F();
    const float* l2w = F(); const float* l2b = F();
    LayP lay[6];
    for (int l = 0; l < 6; ++l) {
        LayP& p = lay[l];
        p.qw = F(); p.qb = F(); p.kw = F(); p.kb = F();
        p.vw = F(); p.vb = F(); p.ow = F(); p.ob = F();
        p.ln1g = F(); p.ln1b = F();
        p.f1w = F(); p.f1b = F(); p.f2w = F(); p.f2b = F();
        p.ln2g = F(); p.ln2b = F();
    }
    const float* r1w = F(); const float* r1b = F();
    const float* r2w = F(); const float* r2b = F();
    (void)l1w;

    // ---------------- workspace carve-up (floats) ----------------
    float* wsf = (float*)d_ws;
    size_t off = 0;
    auto alloc = [&](size_t n) -> float* { float* p = wsf + off; off += n; return p; };
    float* STEM = alloc(25690112);   // stem out; reused as ping buffer after maxpool
    float* X    = alloc(6422528);
    float* T1   = alloc(6422528);
    float* SC   = alloc(3211264);
    float* SEQ  = alloc(180224);
    float* QB   = alloc(180224);
    float* KB   = alloc(180224);
    float* VB   = alloc(180224);
    float* AB   = alloc(180224);
    float* FF1  = alloc(720896);
    float* FF2  = alloc(180224);
    float* GAP  = alloc(16384);
    float* GEMB = alloc(16384);
    float* LOG  = alloc(6272);
    float* LOCC = alloc(512);
    float* POOL = alloc(16384);
    float* RH   = alloc(16384);
    float* RR   = alloc(256);
    int*   IDX  = (int*)alloc(320);

    // ---------------- stem ----------------
    launch_gemm(stream, rgb, stem_w, nullptr,
                stem_bn.g, stem_bn.b, stem_bn.m, stem_bn.v, nullptr, STEM,
                32 * 112 * 112, 64, 3 * 49, 3, 224, 224, 112, 112, 7, 2, 3, 1);
    {
        int total = 32 * 64 * 56 * 56;
        maxpool_kernel<<<(total + 255) / 256, 256, 0, stream>>>(STEM, X, 64, 112, 112, 56, 56);
    }

    // ---------------- residual stages ----------------
    float* cur = X;
    float* oth = STEM;
    int H = 56;
    gi = 0;
    for (int si = 0; si < 4; ++si) {
        for (int bi = 0; bi < nblocks[si]; ++bi) {
            const BlkP& p = blocks[gi++];
            int cin_b = (bi == 0) ? cins[si] : couts[si];
            int cout  = couts[si];
            int st    = (bi == 0) ? strides[si] : 1;
            int Hin   = H;
            int Hout  = Hin / st;
            int M     = 32 * Hout * Hout;
            const float* sc;
            if (p.has) {
                launch_gemm(stream, cur, p.dw, nullptr,
                            p.dbn.g, p.dbn.b, p.dbn.m, p.dbn.v, nullptr, SC,
                            M, cout, cin_b, cin_b, Hin, Hin, Hout, Hout, 1, st, 0, 0);
                sc = SC;
            } else {
                sc = cur;
            }
            launch_gemm(stream, cur, p.w1, nullptr,
                        p.bn1.g, p.bn1.b, p.bn1.m, p.bn1.v, nullptr, T1,
                        M, cout, cin_b * 9, cin_b, Hin, Hin, Hout, Hout, 3, st, 1, 1);
            launch_gemm(stream, T1, p.w2, nullptr,
                        p.bn2.g, p.bn2.b, p.bn2.m, p.bn2.v, sc, oth,
                        M, cout, cout * 9, cout, Hout, Hout, Hout, Hout, 3, 1, 1, 1);
            float* tmp = cur; cur = oth; oth = tmp;
            H = Hout;
        }
        if (si == 2) {
            saliency_kernel<<<(32 * 196 + 255) / 256, 256, 0, stream>>>(cur, salw, salb, LOG);
        }
    }

    // ---------------- global embedding ----------------
    hwmean_kernel<<<(32 * 512 + 255) / 256, 256, 0, stream>>>(cur, GAP, 512, 49);
    launch_gemm(stream, GAP, fcw, fcb, nullptr, nullptr, nullptr, nullptr, nullptr, GEMB,
                32, 512, 512, 0, 0, 0, 0, 0, 0, 0, 0, 0);

    // ---------------- sampling + sequence assembly ----------------
    sample_kernel<<<320, 32, 0, stream>>>(LOG, IDX);
    localconst_kernel<<<1, 512, 0, stream>>>(l1b, l2w, l2b, LOCC);
    seqassemble_kernel<<<(32 * 11 * 512 + 255) / 256, 256, 0, stream>>>(pos, GEMB, LOCC, IDX, SEQ);

    // ---------------- transformer ----------------
    const int Mq = 32 * 11;
    for (int l = 0; l < 6; ++l) {
        const LayP& p = lay[l];
        launch_gemm(stream, SEQ, p.qw, p.qb, nullptr, nullptr, nullptr, nullptr, nullptr, QB,
                    Mq, 512, 512, 0,0,0,0,0, 0,0,0, 0);
        launch_gemm(stream, SEQ, p.kw, p.kb, nullptr, nullptr, nullptr, nullptr, nullptr, KB,
                    Mq, 512, 512, 0,0,0,0,0, 0,0,0, 0);
        launch_gemm(stream, SEQ, p.vw, p.vb, nullptr, nullptr, nullptr, nullptr, nullptr, VB,
                    Mq, 512, 512, 0,0,0,0,0, 0,0,0, 0);
        attn_kernel<<<256, 32, 0, stream>>>(QB, KB, VB, AB);
        launch_gemm(stream, AB, p.ow, p.ob, nullptr, nullptr, nullptr, nullptr, nullptr, QB,
                    Mq, 512, 512, 0,0,0,0,0, 0,0,0, 0);
        lnadd_kernel<<<Mq, 32, 0, stream>>>(SEQ, QB, p.ln1g, p.ln1b);
        launch_gemm(stream, SEQ, p.f1w, p.f1b, nullptr, nullptr, nullptr, nullptr, nullptr, FF1,
                    Mq, 2048, 512, 0,0,0,0,0, 0,0,0, 1);
        launch_gemm(stream, FF1, p.f2w, p.f2b, nullptr, nullptr, nullptr, nullptr, nullptr, FF2,
                    Mq, 512, 2048, 0,0,0,0,0, 0,0,0, 0);
        lnadd_kernel<<<Mq, 32, 0, stream>>>(SEQ, FF2, p.ln2g, p.ln2b);
    }

    // ---------------- rotation head ----------------
    seqmean_kernel<<<(32 * 512 + 255) / 256, 256, 0, stream>>>(SEQ, POOL);
    launch_gemm(stream, POOL, r1w, r1b, nullptr, nullptr, nullptr, nullptr, nullptr, RH,
                32, 512, 512, 0,0,0,0,0, 0,0,0, 1);
    launch_gemm(stream, RH, r2w, r2b, nullptr, nullptr, nullptr, nullptr, nullptr, RR,
                32, 6, 512, 0,0,0,0,0, 0,0,0, 0);
    rot_kernel<<<1, 32, 0, stream>>>(RR, (float*)d_out);
}
